// CosSimConv2D_62818191671383
// MI455X (gfx1250) — matrix operand order repaired and checked by reading, not compiled, not run on hardware
//
#include <hip/hip_runtime.h>

// CosSimConv2D for MI455X (gfx1250, wave32).
// One GEMM (M=B*H*W=131072, K=576, N=128) in bf16 WMMA with fp32 accumulate,
// norms and epilogue in fp32.

#define B_   8
#define H_   128
#define W_   128
#define C_   64
#define U_   128
#define FAN  576           // C*3*3
#define EPSF 1e-12f

typedef __attribute__((ext_vector_type(16))) __bf16 v16bf;
typedef __attribute__((ext_vector_type(8)))  float  v8f;

union Frag16 { v16bf v; uint4 q[2]; };

// round-to-nearest-even f32 -> bf16, packed pair (lo in [15:0], hi in [31:16])
__device__ __forceinline__ unsigned f2bf_pk(float lo, float hi) {
  union { float f; unsigned u; } a, b;
  a.f = lo; b.f = hi;
  unsigned ra = a.u + 0x7FFFu + ((a.u >> 16) & 1u);
  unsigned rb = b.u + 0x7FFFu + ((b.u >> 16) & 1u);
  return (ra >> 16) | (rb & 0xFFFF0000u);
}

// ---------------------------------------------------------------------------
// Prep: qv, softmax(p) exponents, per-unit w-column norm; emit bf16 w/w_norm
// in [unit][k] row-major layout (contiguous k per unit -> B-fragment friendly).
// nb[0..127] = softmax(p); nb[128] = qv. wq = U_ x FAN bf16.
// ---------------------------------------------------------------------------
__global__ __launch_bounds__(128) void cosconv_prep(
    const float* __restrict__ w,   // (FAN, U_) k-major
    const float* __restrict__ p,   // (U_)
    const float* __restrict__ q,   // (1)
    float* __restrict__ nb,
    unsigned short* __restrict__ wq) {
  const int u = threadIdx.x;                 // 0..127
  const float qv = q[0] * q[0] * 0.1f;
  float mx = -3.402823466e38f;
  for (int j = 0; j < U_; ++j) mx = fmaxf(mx, p[j]);
  float s = 0.f;
  for (int j = 0; j < U_; ++j) s += __expf(p[j] - mx);
  nb[u] = __expf(p[u] - mx) / s;
  if (u == 0) nb[U_] = qv;

  float ssq = 0.f;
  for (int k = 0; k < FAN; ++k) {
    float v = w[(size_t)k * U_ + u];
    ssq = fmaf(v, v, ssq);
  }
  const float rwn = 1.f / (sqrtf(fmaxf(ssq, EPSF)) + qv);
  for (int k = 0; k < FAN; k += 2) {
    float v0 = w[(size_t)k * U_ + u] * rwn;
    float v1 = w[(size_t)(k + 1) * U_ + u] * rwn;
    *(unsigned*)&wq[(size_t)u * FAN + k] = f2bf_pk(v0, v1);
  }
}

// ---------------------------------------------------------------------------
// Main: one workgroup per (b,h) image row. 256 threads = 8 waves.
// Wave (wv%4, wv/4) owns a 32(M) x 64(N) tile -> 2x4 v8f accumulators.
// K loop: 18 chunks of 32 (chunk = one 3x3 tap x 32 channels).
// ---------------------------------------------------------------------------
__global__ __launch_bounds__(256) void cosconv_main(
    const float* __restrict__ x,            // (B,H,W,C) f32
    const float* __restrict__ nb,           // e[128], qv
    const unsigned short* __restrict__ wq,  // (U_, FAN) bf16
    float* __restrict__ out) {              // (B,H,W,U_) f32
  __shared__ unsigned short lin[3 * 130 * 64];  // 3 rows, 1-px zero pad each side
  __shared__ float lss[3 * W_];                 // per-pixel channel sum-of-squares
  __shared__ float linv[W_];                    // 1 / x_norm per output pixel

  const int bh  = blockIdx.x;
  const int b   = bh >> 7;
  const int h   = bh & (H_ - 1);
  const int tid = threadIdx.x;

  // ---- stage 3 input rows as bf16 + channel sumsq (fp32) -------------------
  for (int task = tid; task < 3 * W_; task += 256) {
    const int r   = task >> 7;
    const int pix = task & (W_ - 1);
    const int gy  = h + r - 1;
    uint4* dst = (uint4*)&lin[(r * 130 + pix + 1) * 64];
    float ssq = 0.f;
    if (gy >= 0 && gy < H_) {
      const float4* src = (const float4*)&x[(((size_t)b * H_ + gy) * W_ + pix) * C_];
      #pragma unroll
      for (int i = 0; i < 8; ++i) {
        float4 a = src[2 * i], c = src[2 * i + 1];
        ssq += a.x * a.x + a.y * a.y + a.z * a.z + a.w * a.w;
        ssq += c.x * c.x + c.y * c.y + c.z * c.z + c.w * c.w;
        uint4 pk;
        pk.x = f2bf_pk(a.x, a.y); pk.y = f2bf_pk(a.z, a.w);
        pk.z = f2bf_pk(c.x, c.y); pk.w = f2bf_pk(c.z, c.w);
        dst[i] = pk;
      }
    } else {
      uint4 z = {0u, 0u, 0u, 0u};
      #pragma unroll
      for (int i = 0; i < 8; ++i) dst[i] = z;
    }
    lss[r * W_ + pix] = ssq;
  }
  if (tid < 6) {  // zero the left/right padding pixels of each row
    const int r = tid >> 1, side = tid & 1;
    uint4* dst = (uint4*)&lin[(r * 130 + side * 129) * 64];
    uint4 z = {0u, 0u, 0u, 0u};
    #pragma unroll
    for (int i = 0; i < 8; ++i) dst[i] = z;
  }
  __syncthreads();

  if (tid < W_) {  // 1/x_norm from 3x3 window of channel sumsq
    float acc = 0.f;
    #pragma unroll
    for (int dy = 0; dy < 3; ++dy)
      #pragma unroll
      for (int dx = -1; dx <= 1; ++dx) {
        int c = tid + dx;
        if (c >= 0 && c < W_) acc += lss[dy * W_ + c];
      }
    const float qv = nb[U_];
    linv[tid] = 1.f / (sqrtf(fmaxf(acc, EPSF)) + qv);
  }
  __syncthreads();

  // ---- WMMA GEMM -----------------------------------------------------------
  const int wave = tid >> 5, lane = tid & 31;
  const int half = lane >> 4, lm = lane & 15;
  const int mw = (wave & 3) * 32;   // pixel tile base
  const int nw = (wave >> 2) * 64;  // unit  tile base

  v8f acc[2][4] = {};

  for (int kc = 0; kc < 18; ++kc) {
    const int tap = kc >> 1;            // 0..8  (dy,dx tap)
    const int ch0 = (kc & 1) << 5;      // 0 or 32 channel offset within tap
    const int dy  = tap / 3;            // row in padded LDS buffer (0..2)
    const int dx  = tap - 3 * dy;       // padded column offset (0..2)

    // A fragments: 16-bit A 16x32 layout -> lane(lm,half) holds
    // k = 8*half + {0..7} and k = 16 + 8*half + {0..7} (contiguous channels).
    Frag16 A[2];
    #pragma unroll
    for (int mt = 0; mt < 2; ++mt) {
      const unsigned short* base =
          &lin[(dy * 130 + mw + mt * 16 + lm + dx) * 64 + ch0 + 8 * half];
      A[mt].q[0] = *(const uint4*)(base);
      A[mt].q[1] = *(const uint4*)(base + 16);
    }
    // B fragments: lane holds column n = lm, k = kc*32 + 8*half + {0..7} and kc*32 + 16 + 8*half + {0..7}.
    Frag16 Bv[4];
    #pragma unroll
    for (int nt = 0; nt < 4; ++nt) {
      const unsigned short* wb =
          wq + (size_t)(nw + nt * 16 + lm) * FAN + kc * 32 + 8 * half;
      Bv[nt].q[0] = *(const uint4*)(wb);
      Bv[nt].q[1] = *(const uint4*)(wb + 16);
    }
    #pragma unroll
    for (int mt = 0; mt < 2; ++mt)
      #pragma unroll
      for (int nt = 0; nt < 4; ++nt)
        acc[mt][nt] = __builtin_amdgcn_wmma_f32_16x16x32_bf16(
            false, A[mt].v, false, Bv[nt].v, (short)0, acc[mt][nt],
            false, false);
  }

  // ---- epilogue: y = sign(dot) * (|dot/x_norm| + EPS)^softmax(p)[u] --------
  float ev[4];
  #pragma unroll
  for (int nt = 0; nt < 4; ++nt) ev[nt] = nb[nw + nt * 16 + lm];

  float* orow = out + (((size_t)b * H_ + h) * W_) * U_;
  #pragma unroll
  for (int mt = 0; mt < 2; ++mt) {
    const int m0 = mw + mt * 16 + half * 8;  // C/D layout: lanes16-31 -> M=8+r
    const float4 i0 = *(const float4*)&linv[m0];
    const float4 i1 = *(const float4*)&linv[m0 + 4];
    const float inv[8] = {i0.x, i0.y, i0.z, i0.w, i1.x, i1.y, i1.z, i1.w};
    #pragma unroll
    for (int nt = 0; nt < 4; ++nt) {
      const int n = nw + nt * 16 + lm;
      const float e = ev[nt];
      #pragma unroll
      for (int r8 = 0; r8 < 8; ++r8) {
        float y0  = acc[mt][nt][r8] * inv[r8];
        float sgn = (y0 > 0.f) ? 1.f : ((y0 < 0.f) ? -1.f : 0.f);
        float mag = exp2f(e * log2f(fabsf(y0) + EPSF));
        orow[(size_t)(m0 + r8) * U_ + n] = sgn * mag;
      }
    }
  }
}

// ---------------------------------------------------------------------------
extern "C" void kernel_launch(void* const* d_in, const int* in_sizes, int n_in,
                              void* d_out, int out_size, void* d_ws, size_t ws_size,
                              hipStream_t stream) {
  (void)in_sizes; (void)n_in; (void)out_size; (void)ws_size;
  const float* x = (const float*)d_in[0];   // (B,H,W,C)
  const float* w = (const float*)d_in[1];   // (1,FAN,U)
  const float* p = (const float*)d_in[2];   // (U)
  const float* q = (const float*)d_in[3];   // (1)

  float*          nb = (float*)d_ws;                               // e[128], qv
  unsigned short* wq = (unsigned short*)((char*)d_ws + 1024);      // U x FAN bf16

  cosconv_prep<<<1, 128, 0, stream>>>(w, p, q, nb, wq);
  cosconv_main<<<B_ * H_, 256, 0, stream>>>(x, nb, wq, (float*)d_out);
}
